// HierarchicalMoE_91096256348839
// MI455X (gfx1250) — matrix operand order repaired and checked by reading, not compiled, not run on hardware
//
#include <hip/hip_runtime.h>
#include <cstdint>

// ---------------- problem constants ----------------
#define NTOK 2048
#define DIN  1024
#define HID  4096
#define OOUT 1024
#define NG   4
#define NE   4
#define NZ   16          // NG*NE experts
#define EPSF 2.2204460492503131e-16f

typedef __bf16 bf16_t;
typedef __attribute__((ext_vector_type(16))) __bf16 v16bf;
typedef __attribute__((ext_vector_type(8)))  __bf16 v8bf;
typedef __attribute__((ext_vector_type(8)))  float  v8f;
typedef __attribute__((ext_vector_type(4)))  unsigned int u32x4;
typedef __attribute__((ext_vector_type(8)))  int i32x8;
typedef __attribute__((ext_vector_type(4)))  int i32x4;

__device__ __forceinline__ bf16_t f2bf(float f) {
    union { float f; uint32_t u; } v; v.f = f;
    uint32_t r = v.u + 0x7FFFu + ((v.u >> 16) & 1u);   // RNE
    uint16_t h = (uint16_t)(r >> 16);
    return __builtin_bit_cast(bf16_t, h);
}

#define SHUF16(lo, hi) __builtin_shufflevector((lo), (hi), 0,1,2,3,4,5,6,7,8,9,10,11,12,13,14,15)

// ---------------- elementwise fp32 -> bf16 ----------------
__global__ void cvt_bf16_kernel(const float* __restrict__ src, bf16_t* __restrict__ dst, int n) {
    int i = blockIdx.x * 256 + threadIdx.x;
    if (i < n) dst[i] = f2bf(src[i]);
}

// ---------------- transpose + convert: src [Z][R][C] fp32 -> dst [Z][C][R] bf16 ----------------
__global__ void transcvt_kernel(const float* __restrict__ src, bf16_t* __restrict__ dst,
                                int R, int C) {
    __shared__ float tile[32][33];
    int z  = blockIdx.z;
    const float* s = src + (size_t)z * R * C;
    bf16_t*      d = dst + (size_t)z * R * C;
    int c0 = blockIdx.x * 32, r0 = blockIdx.y * 32;
    int tx = threadIdx.x, ty = threadIdx.y;
    #pragma unroll
    for (int r = ty; r < 32; r += 8)
        tile[r][tx] = s[(size_t)(r0 + r) * C + c0 + tx];
    __syncthreads();
    #pragma unroll
    for (int r = ty; r < 32; r += 8)
        d[(size_t)(c0 + r) * R + r0 + tx] = f2bf(tile[tx][r]);
}

// ---------------- gating: outer [N,G] + inner [N,G,E] top-2 softmax gates ----------------
__device__ __forceinline__ void top2_softmax4(const float* v, float* out) {
    int i1 = 0;
    #pragma unroll
    for (int i = 1; i < 4; ++i) if (v[i] > v[i1]) i1 = i;
    int i2 = -1;
    #pragma unroll
    for (int i = 0; i < 4; ++i) {
        if (i == i1) continue;
        if (i2 < 0 || v[i] > v[i2]) i2 = i;
    }
    float e2 = expf(v[i2] - v[i1]);
    float s  = 1.0f + e2;
    #pragma unroll
    for (int i = 0; i < 4; ++i) out[i] = 0.0f;
    out[i1] = 1.0f / s;
    out[i2] = e2 / s;
}

__global__ void gate_kernel(const float* __restrict__ x, const float* __restrict__ wgo,
                            const float* __restrict__ wgi,
                            float* __restrict__ og, float* __restrict__ ig) {
    int n = blockIdx.x, tid = threadIdx.x;
    float p[20];
    #pragma unroll
    for (int j = 0; j < 20; ++j) p[j] = 0.0f;
    const float* xr = x + (size_t)n * DIN;
    for (int d = tid; d < DIN; d += 256) {
        float xv = xr[d];
        #pragma unroll
        for (int j = 0; j < 4; ++j) p[j] += xv * wgo[d * NG + j];
        #pragma unroll
        for (int g = 0; g < 4; ++g)
            #pragma unroll
            for (int e = 0; e < 4; ++e)
                p[4 + g * 4 + e] += xv * wgi[g * (DIN * NE) + d * NE + e];
    }
    #pragma unroll
    for (int j = 0; j < 20; ++j) {
        float v = p[j];
        #pragma unroll
        for (int off = 16; off; off >>= 1) v += __shfl_xor(v, off, 32);
        p[j] = v;
    }
    __shared__ float sbuf[20][8];
    int wave = tid >> 5, lane = tid & 31;
    if (lane == 0) {
        #pragma unroll
        for (int j = 0; j < 20; ++j) sbuf[j][wave] = p[j];
    }
    __syncthreads();
    if (tid == 0) {
        float t[20];
        #pragma unroll
        for (int j = 0; j < 20; ++j) {
            float s = 0.0f;
            #pragma unroll
            for (int w = 0; w < 8; ++w) s += sbuf[j][w];
            t[j] = s;
        }
        float gv[4];
        top2_softmax4(t, gv);
        #pragma unroll
        for (int j = 0; j < 4; ++j) og[n * NG + j] = gv[j];
        #pragma unroll
        for (int g = 0; g < 4; ++g) {
            top2_softmax4(t + 4 + g * 4, gv);
            #pragma unroll
            for (int e = 0; e < 4; ++e) ig[n * NZ + g * 4 + e] = gv[e];
        }
    }
}

// ---------------- Tensor Data Mover tile fetch ----------------
#define BM 128
#define BN 128
#define BK 32
#define LDP 40   // padded LDS row (bf16 elems): 80B stride -> conflict-free b128 frag loads

#if __has_builtin(__builtin_amdgcn_tensor_load_to_lds)
#define USE_TDM 1
#else
#define USE_TDM 0
#endif

#if USE_TDM
// Load a rows x BK bf16 tile (row stride ld_elems) into LDS at lds_byte_off,
// with 16B padding inserted after each 64B row (matches LDP=40 layout).
// D# bitfields per CDNA5 ISA sec 8.3/8.4.
__device__ __forceinline__ void tdm_tile_load(const bf16_t* gsrc, unsigned lds_byte_off,
                                              int rows, int ld_elems) {
    unsigned long long ga = (unsigned long long)gsrc;
    u32x4 g0;
    g0.x = 1u;                                                  // count=1, user load desc
    g0.y = lds_byte_off;                                        // lds_addr (bytes)
    g0.z = (unsigned)ga;                                        // global_addr[31:0]
    g0.w = (unsigned)((ga >> 32) & 0x01FFFFFFu) | (2u << 30);   // global_addr[56:32] | type=2
    unsigned dim0 = (unsigned)ld_elems;                         // tensor dim0 (elements)
    unsigned dim1 = (unsigned)rows;                             // tensor dim1
    i32x8 g1;
    g1[0] = (int)((1u << 16)        // data_size = 2 bytes
                | (1u << 20)        // pad_enable
                | (3u << 22)        // pad_interval = 16 DWORDs (one 64B row)
                | (3u << 25));      // pad_amount   = 4 DWORDs (16B)
    g1[1] = (int)((dim0 & 0xFFFFu) << 16);                      // tensor_dim0[15:0]
    g1[2] = (int)((dim0 >> 16) | ((dim1 & 0xFFFFu) << 16));     // dim0[31:16] | dim1[15:0]
    g1[3] = (int)((dim1 >> 16) | ((unsigned)BK << 16));         // dim1[31:16] | tile_dim0
    g1[4] = rows;                                               // tile_dim1 (tile_dim2 = 0)
    g1[5] = ld_elems;                                           // tensor_dim0_stride[31:0]
    g1[6] = 0;
    g1[7] = 0;
    i32x4 gz = {0, 0, 0, 0};
#if defined(__clang_major__) && (__clang_major__ >= 23)
    i32x8 gz8 = {0, 0, 0, 0, 0, 0, 0, 0};
    __builtin_amdgcn_tensor_load_to_lds(g0, g1, gz, gz, gz8, 0);
#else
    __builtin_amdgcn_tensor_load_to_lds(g0, g1, gz, gz, 0);
#endif
}
#endif

// LDS region offsets (bf16 elements) within smem (A double buffer, then B double buffer)
#define SM_A(buf)  ((buf) * (BM * LDP))
#define SM_B(buf)  (2 * (BM * LDP) + (buf) * (BN * LDP))

// ---------------- bf16 WMMA GEMM: C[z] = act(A[z][M,K] * B[z][Nn,K]^T + bias[z]) ----------------
template <bool GELU>
__global__ __launch_bounds__(256) void gemm_kernel(
    const bf16_t* __restrict__ A, size_t sAz,
    const bf16_t* __restrict__ B, size_t sBz,
    const float*  __restrict__ bias, int Nn,
    bf16_t* __restrict__ Cb, float* __restrict__ Cf, size_t sCz, int K) {

    __shared__ alignas(16) bf16_t smem[2 * BM * LDP + 2 * BN * LDP];   // 40 KB

    const int z  = blockIdx.z;
    const int n0 = blockIdx.x * BN;
    const int m0 = blockIdx.y * BM;
    const bf16_t* Ag = A + (size_t)z * sAz + (size_t)m0 * K;
    const bf16_t* Bg = B + (size_t)z * sBz + (size_t)n0 * K;
    const float* biasz = bias + (size_t)z * Nn;

    const int tid  = threadIdx.x;
    const int wave = tid >> 5;
    const int lane = tid & 31;
    const int wm   = wave & 3;   // 4 waves along M -> 32 rows each
    const int wn   = wave >> 2;  // 2 waves along N -> 64 cols each
    const int half = lane >> 4;
    const int l16  = lane & 15;
    const int nk   = K / BK;

#if USE_TDM
    // LDS aperture: generic-pointer low 32 bits == LDS byte offset (ISA 10.2).
    // Single runtime addrspacecast (no pointer-array static initializers).
    const unsigned ldsbase = (unsigned)(unsigned long long)(uintptr_t)&smem[0];
    const unsigned offA0 = ldsbase + SM_A(0) * 2;
    const unsigned offA1 = ldsbase + SM_A(1) * 2;
    const unsigned offB0 = ldsbase + SM_B(0) * 2;
    const unsigned offB1 = ldsbase + SM_B(1) * 2;
    if (tid < 32) {                       // wave 0 drives the Tensor Data Mover
        tdm_tile_load(Ag, offA0, BM, K);
        tdm_tile_load(Bg, offB0, BN, K);
        __builtin_amdgcn_s_wait_tensorcnt(0);
    }
    __syncthreads();
#else
    // fallback: register-staged global->LDS (both tiles are 128 rows x 32 halves)
    const int ar = tid >> 1, ac = (tid & 1) * 16;
    const bf16_t* Aptr = Ag + (size_t)ar * K + ac;
    const bf16_t* Bptr = Bg + (size_t)ar * K + ac;
    uint4 ra0, ra1, rb0, rb1;
    {
        const uint4* ap = (const uint4*)(Aptr);
        ra0 = ap[0]; ra1 = ap[1];
        const uint4* bp = (const uint4*)(Bptr);
        rb0 = bp[0]; rb1 = bp[1];
        uint4* as = (uint4*)(&smem[SM_A(0) + ar * LDP + ac]);
        as[0] = ra0; as[1] = ra1;
        uint4* bs = (uint4*)(&smem[SM_B(0) + ar * LDP + ac]);
        bs[0] = rb0; bs[1] = rb1;
    }
    __syncthreads();
#endif

    v8f acc[2][4];
    {
        v8f zero = {0.f, 0.f, 0.f, 0.f, 0.f, 0.f, 0.f, 0.f};
        #pragma unroll
        for (int i = 0; i < 2; ++i)
            #pragma unroll
            for (int j = 0; j < 4; ++j) acc[i][j] = zero;
    }

    for (int kt = 0; kt < nk; ++kt) {
        const int cur = kt & 1;
#if USE_TDM
        if (kt + 1 < nk && tid < 32) {    // DMA next tile while computing this one
            tdm_tile_load(Ag + (kt + 1) * BK, cur ? offA0 : offA1, BM, K);
            tdm_tile_load(Bg + (kt + 1) * BK, cur ? offB0 : offB1, BN, K);
        }
#else
        if (kt + 1 < nk) {
            const uint4* ap = (const uint4*)(Aptr + (kt + 1) * BK);
            ra0 = ap[0]; ra1 = ap[1];
            const uint4* bp = (const uint4*)(Bptr + (kt + 1) * BK);
            rb0 = bp[0]; rb1 = bp[1];
        }
#endif
        const bf16_t* Ab = smem + SM_A(0) + (cur ? BM * LDP : 0);
        const bf16_t* Bb = smem + SM_B(0) + (cur ? BN * LDP : 0);

        v16bf afrag[2], bfrag[4];
        #pragma unroll
        for (int t = 0; t < 2; ++t) {
            const bf16_t* p = Ab + (wm * 32 + t * 16 + l16) * LDP + half * 8;
            v8bf lo = *(const v8bf*)p;
            v8bf hi = *(const v8bf*)(p + 16);
            afrag[t] = SHUF16(lo, hi);
        }
        #pragma unroll
        for (int t = 0; t < 4; ++t) {
            const bf16_t* p = Bb + (wn * 64 + t * 16 + l16) * LDP + half * 8;
            v8bf lo = *(const v8bf*)p;
            v8bf hi = *(const v8bf*)(p + 16);
            bfrag[t] = SHUF16(lo, hi);
        }
        #pragma unroll
        for (int i = 0; i < 2; ++i)
            #pragma unroll
            for (int j = 0; j < 4; ++j)
                acc[i][j] = __builtin_amdgcn_wmma_f32_16x16x32_bf16(
                    false, afrag[i], false, bfrag[j], (short)0, acc[i][j], false, false);

        if (kt + 1 < nk) {
#if USE_TDM
            if (tid < 32) __builtin_amdgcn_s_wait_tensorcnt(0);
            __syncthreads();
#else
            uint4* as = (uint4*)(&smem[SM_A(0) + (cur ? 0 : BM * LDP) + ar * LDP + ac]);
            as[0] = ra0; as[1] = ra1;
            uint4* bs = (uint4*)(&smem[SM_B(0) + (cur ? 0 : BN * LDP) + ar * LDP + ac]);
            bs[0] = rb0; bs[1] = rb1;
            __syncthreads();
#endif
        }
    }

    // epilogue
    bf16_t* Cbz = Cb ? Cb + (size_t)z * sCz : nullptr;
    float*  Cfz = Cf ? Cf + (size_t)z * sCz : nullptr;
    #pragma unroll
    for (int i = 0; i < 2; ++i) {
        int mbase = m0 + wm * 32 + i * 16 + half * 8;
        #pragma unroll
        for (int j = 0; j < 4; ++j) {
            int col = n0 + wn * 64 + j * 16 + l16;
            float bv = biasz[col];
            #pragma unroll
            for (int r = 0; r < 8; ++r) {
                float v = acc[i][j][r] + bv;
                size_t idx = (size_t)(mbase + r) * Nn + col;
                if (GELU) {
                    float g = 0.5f * v * (1.0f + erff(v * 0.70710678118654752440f));
                    Cbz[idx] = f2bf(g);
                } else {
                    Cfz[idx] = v;
                }
            }
        }
    }
}

// ---------------- per-row logsumexp over O=1024 ----------------
__global__ void lse_kernel(const float* __restrict__ outf, float* __restrict__ lse) {
    const float* r = outf + (size_t)blockIdx.x * OOUT;
    int tid = threadIdx.x;
    float vals[4], m = -3.4e38f;
    #pragma unroll
    for (int i = 0; i < 4; ++i) { vals[i] = r[tid + i * 256]; m = fmaxf(m, vals[i]); }
    #pragma unroll
    for (int off = 16; off; off >>= 1) m = fmaxf(m, __shfl_xor(m, off, 32));
    __shared__ float sm[8], ss[8], sM;
    int wave = tid >> 5, lane = tid & 31;
    if (lane == 0) sm[wave] = m;
    __syncthreads();
    if (tid == 0) {
        float mm = sm[0];
        #pragma unroll
        for (int w = 1; w < 8; ++w) mm = fmaxf(mm, sm[w]);
        sM = mm;
    }
    __syncthreads();
    float M = sM, s = 0.0f;
    #pragma unroll
    for (int i = 0; i < 4; ++i) s += expf(vals[i] - M);
    #pragma unroll
    for (int off = 16; off; off >>= 1) s += __shfl_xor(s, off, 32);
    if (lane == 0) ss[wave] = s;
    __syncthreads();
    if (tid == 0) {
        float t = 0.0f;
        #pragma unroll
        for (int w = 0; w < 8; ++w) t += ss[w];
        lse[blockIdx.x] = M + logf(t);
    }
}

// ---------------- hierarchical combine ----------------
__global__ void combine_kernel(const float* __restrict__ outf, const float* __restrict__ lse,
                               const float* __restrict__ og, const float* __restrict__ ig,
                               float* __restrict__ out) {
    int n = blockIdx.y;
    int o = blockIdx.x * 256 + threadIdx.x;
    float acc2 = 0.0f;
    #pragma unroll
    for (int g = 0; g < 4; ++g) {
        float comb = 0.0f;
        #pragma unroll
        for (int e = 0; e < 4; ++e) {
            int z = g * 4 + e;
            float gate = ig[n * NZ + z];
            float lp = outf[((size_t)z * NTOK + n) * OOUT + o] - lse[z * NTOK + n];
            comb += gate * expf(lp);
        }
        if (comb == 0.0f) comb = EPSF;
        acc2 += og[n * NG + g] * comb;
    }
    if (acc2 == 0.0f) acc2 = EPSF;
    out[(size_t)n * OOUT + o] = logf(acc2);
}

// ---------------- workspace layout (bytes) ----------------
static constexpr size_t OFF_XB  = 0;                                    // N*D bf16     =   4 MB
static constexpr size_t OFF_W1B = OFF_XB  + (size_t)NTOK * DIN * 2;     // Z*H*D bf16   = 128 MB
static constexpr size_t OFF_W2B = OFF_W1B + (size_t)NZ * HID * DIN * 2; // Z*O*H bf16   = 128 MB
static constexpr size_t OFF_HB  = OFF_W2B + (size_t)NZ * OOUT * HID * 2;// Z*N*H bf16   = 256 MB
static constexpr size_t OFF_OUT = OFF_HB  + (size_t)NZ * NTOK * HID * 2;// Z*N*O f32    = 128 MB
static constexpr size_t OFF_OG  = OFF_OUT + (size_t)NZ * NTOK * OOUT * 4;
static constexpr size_t OFF_IG  = OFF_OG  + (size_t)NTOK * NG * 4;
static constexpr size_t OFF_LSE = OFF_IG  + (size_t)NTOK * NZ * 4;

extern "C" void kernel_launch(void* const* d_in, const int* in_sizes, int n_in,
                              void* d_out, int out_size, void* d_ws, size_t ws_size,
                              hipStream_t stream) {
    const float* x   = (const float*)d_in[0];
    const float* wgo = (const float*)d_in[1];
    const float* wgi = (const float*)d_in[2];
    const float* W1  = (const float*)d_in[3];
    const float* b1  = (const float*)d_in[4];
    const float* W2  = (const float*)d_in[5];
    const float* b2  = (const float*)d_in[6];
    float* out = (float*)d_out;

    char* ws = (char*)d_ws;
    bf16_t* xb   = (bf16_t*)(ws + OFF_XB);
    bf16_t* w1b  = (bf16_t*)(ws + OFF_W1B);
    bf16_t* w2b  = (bf16_t*)(ws + OFF_W2B);
    bf16_t* hb   = (bf16_t*)(ws + OFF_HB);
    float*  outf = (float*)(ws + OFF_OUT);
    float*  og   = (float*)(ws + OFF_OG);
    float*  ig   = (float*)(ws + OFF_IG);
    float*  lseb = (float*)(ws + OFF_LSE);

    // 1) precision conversion (+ weight transpose to N-major for WMMA-friendly tiles)
    cvt_bf16_kernel<<<(NTOK * DIN + 255) / 256, 256, 0, stream>>>(x, xb, NTOK * DIN);
    transcvt_kernel<<<dim3(HID / 32, DIN / 32, NZ), dim3(32, 8), 0, stream>>>(W1, w1b, DIN, HID);
    transcvt_kernel<<<dim3(OOUT / 32, HID / 32, NZ), dim3(32, 8), 0, stream>>>(W2, w2b, HID, OOUT);

    // 2) gating (outer + inner top-2 softmax)
    gate_kernel<<<NTOK, 256, 0, stream>>>(x, wgo, wgi, og, ig);

    // 3) fc1 + exact gelu  -> hb (bf16)
    gemm_kernel<true><<<dim3(HID / BN, NTOK / BM, NZ), 256, 0, stream>>>(
        xb, (size_t)0, w1b, (size_t)HID * DIN, b1, HID,
        hb, nullptr, (size_t)NTOK * HID, DIN);

    // 4) fc2 -> outf (f32 logits)
    gemm_kernel<false><<<dim3(OOUT / BN, NTOK / BM, NZ), 256, 0, stream>>>(
        hb, (size_t)NTOK * HID, w2b, (size_t)OOUT * HID, b2, OOUT,
        nullptr, outf, (size_t)NTOK * OOUT, HID);

    // 5) log-softmax normalizers + hierarchical combine
    lse_kernel<<<NZ * NTOK, 256, 0, stream>>>(outf, lseb);
    combine_kernel<<<dim3(OOUT / 256, NTOK), 256, 0, stream>>>(outf, lseb, og, ig, out);
}
